// VisitTransformer_52347061403935
// MI455X (gfx1250) — matrix operand order repaired and checked by reading, not compiled, not run on hardware
//
#include <hip/hip_runtime.h>
#include <hip/hip_bf16.h>
#include <math.h>

// ---------------------------------------------------------------------------
// VisitTransformer forward for gfx1250 (MI455X).
// bf16 WMMA (v_wmma_f32_16x16x32_bf16) for all matmuls, fp32 accumulate,
// fp32 residual/LayerNorm path.
// ---------------------------------------------------------------------------

typedef __attribute__((ext_vector_type(16))) __bf16 v16bf;
typedef __attribute__((ext_vector_type(8)))  float  v8f;

#define BATCH   128
#define SEQ     510
#define SEQP    512
#define EMB     256
#define HEADS   2
#define DHEAD   128
#define FFDIM   1024
#define NLAY    2
#define MROWS   (BATCH*SEQ)     // 65280 rows, divisible by 128

// ---- WMMA fragment loaders (layouts per CDNA5 ISA 7.12.2) ------------------
// A (16x32, 16-bit): lane L holds row M=L%16; lo-half K = (L<16? 0..7 : 8..15),
// hi-half K = +16.  Two contiguous 16B chunks per lane.
__device__ __forceinline__ v16bf load_a_frag(const __bf16* p) {
  v16bf a;
  uint4* ap = reinterpret_cast<uint4*>(&a);
  ap[0] = *reinterpret_cast<const uint4*>(p);
  ap[1] = *reinterpret_cast<const uint4*>(p + 16);
  return a;
}
// B (32x16, 16-bit): lane L holds col N=L%16; lo-half K = (L<16? 0..7 : 8..15),
// hi-half K = +16 (source must be K-contiguous = B^T).
__device__ __forceinline__ v16bf load_b_frag(const __bf16* p) {
  v16bf b;
  uint4* bp = reinterpret_cast<uint4*>(&b);
  bp[0] = *reinterpret_cast<const uint4*>(p);
  bp[1] = *reinterpret_cast<const uint4*>(p + 16);
  return b;
}

// ---- weight prep: fp32 [K,N] -> bf16 transposed [N,K] ----------------------
__global__ void wprep_kernel(const float* __restrict__ W, __bf16* __restrict__ Wt,
                             int K, int N) {
  int idx = blockIdx.x * blockDim.x + threadIdx.x;
  if (idx >= K * N) return;
  int k = idx / N, n = idx - k * N;
  Wt[(size_t)n * K + k] = (__bf16)W[idx];
}

// ---- fill padded x with pad_emb -------------------------------------------
__global__ void fill_pad_kernel(float* __restrict__ x, const float* __restrict__ pemb) {
  size_t i = (size_t)blockIdx.x * blockDim.x + threadIdx.x;
  if (i < (size_t)MROWS * EMB) x[i] = pemb[i & (EMB - 1)];
}

// ---- per-visit embedding: sum 16 code embeddings + sinusoidal time emb -----
__global__ void visit_embed_kernel(const int* __restrict__ codes,
                                   const int* __restrict__ dest,
                                   const float* __restrict__ times,
                                   const float* __restrict__ cemb,
                                   float* __restrict__ x) {
  int v = blockIdx.x;            // 0 .. B*V-1 (16384)
  int d = threadIdx.x;           // 0 .. 255
  const int* cp = codes + v * 16;
  float s = 0.f;
  #pragma unroll
  for (int j = 0; j < 16; ++j) s += cemb[(size_t)cp[j] * EMB + d];
  float t = fminf(fmaxf(times[v], 0.f), 364.f);
  const float lr = -9.2103403719761836f / 256.0f;   // ln(1e-4)/D
  float te = (d < 128) ? __sinf(t * __expf((float)d * lr))
                       : __cosf(t * __expf((float)(d - 128) * lr));
  x[(size_t)dest[v] * EMB + d] = s + te;
}

// ---- fp32 -> bf16 ----------------------------------------------------------
__global__ void cvt_bf16_kernel(const float* __restrict__ src, __bf16* __restrict__ dst,
                                size_t n) {
  size_t i = (size_t)blockIdx.x * blockDim.x + threadIdx.x;
  if (i < n) dst[i] = (__bf16)src[i];
}

// ---- V [B*S, H*DH] bf16 -> Vt [B,H,DH,SEQP] bf16 (zero tail) ---------------
__global__ void vtrans_kernel(const __bf16* __restrict__ vsrc, __bf16* __restrict__ vt) {
  size_t i = (size_t)blockIdx.x * blockDim.x + threadIdx.x;
  if (i >= (size_t)BATCH * HEADS * DHEAD * SEQP) return;
  int sp = (int)(i & (SEQP - 1));
  size_t r = i >> 9;
  int dh = (int)(r & (DHEAD - 1)); r >>= 7;
  int h  = (int)(r & 1);
  int b  = (int)(r >> 1);
  __bf16 val = (__bf16)0.f;
  if (sp < SEQ) val = vsrc[((size_t)(b * SEQ + sp)) * EMB + h * DHEAD + dh];
  vt[i] = val;
}

// ---- generic bf16 WMMA GEMM: out = act( A[M,K] @ Bt[N,K]^T + bias + res ) ---
// block 256 threads = 8 waves, block tile 128x128, wave tile 32x64.
__global__ void gemm_bf16_kernel(const __bf16* __restrict__ A,
                                 const __bf16* __restrict__ Bt,
                                 const float* __restrict__ bias,
                                 const float* __restrict__ res,
                                 float* __restrict__ outF,
                                 __bf16* __restrict__ outB,
                                 int M, int N, int K, int relu) {
  const int lane = threadIdx.x & 31;
  const int wave = threadIdx.x >> 5;
  const int wy = wave & 3;           // 4 row waves -> 128 rows
  const int wx = wave >> 2;          // 2 col waves -> 128 cols
  const int m0 = blockIdx.y * 128 + wy * 32;
  const int n0 = blockIdx.x * 128 + wx * 64;
  const int lr = lane & 15;
  const int hi = lane >> 4;

  v8f acc[2][4] = {};

  for (int kk = 0; kk < K; kk += 32) {
    v16bf af[2], bfr[4];
    #pragma unroll
    for (int r = 0; r < 2; ++r) {
      const __bf16* p = A + (size_t)(m0 + r * 16 + lr) * K + kk + hi * 8;
      __builtin_prefetch(p + 64, 0, 1);       // global_prefetch_b8 on A stream
      af[r] = load_a_frag(p);
    }
    #pragma unroll
    for (int c = 0; c < 4; ++c)
      bfr[c] = load_b_frag(Bt + (size_t)(n0 + c * 16 + lr) * K + kk + hi * 8);
    #pragma unroll
    for (int r = 0; r < 2; ++r)
      #pragma unroll
      for (int c = 0; c < 4; ++c)
        acc[r][c] = __builtin_amdgcn_wmma_f32_16x16x32_bf16(
            false, af[r], false, bfr[c], (short)0, acc[r][c], false, false);
  }

  // C layout: VGPR i -> row m0 + i + (lane>=16 ? 8 : 0), col n0 + lane%16.
  #pragma unroll
  for (int r = 0; r < 2; ++r) {
    #pragma unroll
    for (int c = 0; c < 4; ++c) {
      #pragma unroll
      for (int i = 0; i < 8; ++i) {
        int row = m0 + r * 16 + hi * 8 + i;
        int col = n0 + c * 16 + lr;
        float v = acc[r][c][i];
        if (bias) v += bias[col];
        if (res)  v += res[(size_t)row * N + col];
        if (relu) v = v > 0.f ? v : 0.f;
        if (outF) outF[(size_t)row * N + col] = v;
        if (outB) outB[(size_t)row * N + col] = (__bf16)v;
      }
    }
  }
}

// ---- attention: softmax(Q K^T / sqrt(DH)) @ V, full 510-token context ------
// grid = B*H*16, block = 64 (2 waves), each wave owns 16 query rows.
// dyn LDS: 2 waves * 16 rows * 520 floats (520 => bank-conflict-free rows).
#define SROW 520
__global__ void attn_kernel(const __bf16* __restrict__ qb,
                            const __bf16* __restrict__ kb,
                            const __bf16* __restrict__ vt,
                            __bf16* __restrict__ cb) {
  extern __shared__ float smem[];
  const int lane = threadIdx.x & 31;
  const int wave = threadIdx.x >> 5;
  float* sc = smem + wave * 16 * SROW;

  int bid  = blockIdx.x;
  int qblk = bid & 15; bid >>= 4;
  int h = bid & 1;
  int b = bid >> 1;
  int q0 = qblk * 32 + wave * 16;
  const int lr = lane & 15;
  const int hi = lane >> 4;

  // Q fragments: DH=128 -> 4 K-chunks of 32
  int qrow = q0 + lr; if (qrow > SEQ - 1) qrow = SEQ - 1;
  const __bf16* qbase = qb + ((size_t)(b * SEQ + qrow)) * EMB + h * DHEAD + hi * 8;
  v16bf qa[4];
  #pragma unroll
  for (int ks = 0; ks < 4; ++ks) qa[ks] = load_a_frag(qbase + ks * 32);

  const float scale = 0.08838834764831845f;   // 1/sqrt(128)

  // scores -> LDS (padded keys 510..511 get -inf)
  for (int kt = 0; kt < 32; ++kt) {
    int key = kt * 16 + lr;
    int keyc = key > SEQ - 1 ? SEQ - 1 : key;
    const __bf16* kbase = kb + ((size_t)(b * SEQ + keyc)) * EMB + h * DHEAD + hi * 8;
    v8f s = {};
    #pragma unroll
    for (int ks = 0; ks < 4; ++ks) {
      v16bf bfr = load_b_frag(kbase + ks * 32);
      s = __builtin_amdgcn_wmma_f32_16x16x32_bf16(false, qa[ks], false, bfr,
                                                  (short)0, s, false, false);
    }
    bool valid = key < SEQ;
    #pragma unroll
    for (int i = 0; i < 8; ++i) {
      int m = hi * 8 + i;
      sc[m * SROW + kt * 16 + lr] = valid ? s[i] * scale : -1e30f;
    }
  }
  __syncthreads();

  // softmax stats: lane L scans half (L/16) of row (L%16), combine via shfl_xor
  const float* rp = sc + (lane & 15) * SROW + hi * 256;
  float mx = -1e30f;
  for (int k = 0; k < 256; ++k) mx = fmaxf(mx, rp[k]);
  mx = fmaxf(mx, __shfl_xor(mx, 16));
  float sum = 0.f;
  for (int k = 0; k < 256; ++k) sum += __expf(rp[k] - mx);
  sum += __shfl_xor(sum, 16);
  float recip = 1.f / sum;

  // ctx = exp(scores - mx) @ V ; exp materialized straight into A-fragments
  v8f acc[8] = {};
  const float* arow = sc + (lane & 15) * SROW;
  const __bf16* vhead = vt + ((size_t)(b * HEADS + h)) * DHEAD * SEQP;
  for (int kt2 = 0; kt2 < 16; ++kt2) {
    int kb0 = kt2 * 32 + hi * 8;
    v16bf pa;
    #pragma unroll
    for (int j = 0; j < 8; ++j) pa[j]     = (__bf16)__expf(arow[kb0 + j] - mx);
    #pragma unroll
    for (int j = 0; j < 8; ++j) pa[8 + j] = (__bf16)__expf(arow[kb0 + 16 + j] - mx);
    const __bf16* vbase = vhead + kt2 * 32 + hi * 8;
    #pragma unroll
    for (int c = 0; c < 8; ++c) {
      v16bf bfr = load_b_frag(vbase + (size_t)(c * 16 + lr) * SEQP);
      acc[c] = __builtin_amdgcn_wmma_f32_16x16x32_bf16(false, pa, false, bfr,
                                                       (short)0, acc[c], false, false);
    }
  }

  // epilogue: fold 1/sum per row, store bf16 ctx
  #pragma unroll
  for (int i = 0; i < 8; ++i) {
    float f0 = __shfl(recip, i);
    float f1 = __shfl(recip, i + 8);
    float f  = hi ? f1 : f0;
    int row = q0 + hi * 8 + i;
    if (row < SEQ) {
      #pragma unroll
      for (int c = 0; c < 8; ++c)
        cb[((size_t)(b * SEQ + row)) * EMB + h * DHEAD + c * 16 + lr] =
            (__bf16)(acc[c][i] * f);
    }
  }
}

// ---- LayerNorm over D=256: one wave per row, 8 rows per 256-thread block ---
__global__ void ln_kernel(const float* __restrict__ src,
                          const float* __restrict__ g, const float* __restrict__ bta,
                          float* __restrict__ outF, __bf16* __restrict__ outB) {
  int wave = threadIdx.x >> 5;
  int lane = threadIdx.x & 31;
  size_t row = (size_t)blockIdx.x * 8 + wave;
  const float* rp = src + row * EMB;
  float v[8], s = 0.f, s2 = 0.f;
  #pragma unroll
  for (int j = 0; j < 8; ++j) {
    v[j] = rp[lane + j * 32];
    s += v[j]; s2 += v[j] * v[j];
  }
  #pragma unroll
  for (int o = 16; o > 0; o >>= 1) { s += __shfl_xor(s, o); s2 += __shfl_xor(s2, o); }
  float mean = s * (1.f / EMB);
  float var  = s2 * (1.f / EMB) - mean * mean;
  float rs   = rsqrtf(var + 1e-5f);
  #pragma unroll
  for (int j = 0; j < 8; ++j) {
    int col = lane + j * 32;
    float y = g[col] * (v[j] - mean) * rs + bta[col];
    outF[row * EMB + col] = y;
    if (outB) outB[row * EMB + col] = (__bf16)y;
  }
}

// ---------------------------------------------------------------------------
extern "C" void kernel_launch(void* const* d_in, const int* in_sizes, int n_in,
                              void* d_out, int out_size, void* d_ws, size_t ws_size,
                              hipStream_t stream) {
  (void)in_sizes; (void)n_in; (void)out_size; (void)ws_size;
  const int*   codes = (const int*)d_in[0];
  const int*   dest  = (const int*)d_in[2];
  const float* times = (const float*)d_in[3];
  const float* cemb  = (const float*)d_in[4];
  const float* pemb  = (const float*)d_in[5];
  const float* Wq = (const float*)d_in[6];
  const float* Wk = (const float*)d_in[7];
  const float* Wv = (const float*)d_in[8];
  const float* Wo = (const float*)d_in[9];
  const float* bq = (const float*)d_in[10];
  const float* bk = (const float*)d_in[11];
  const float* bv = (const float*)d_in[12];
  const float* bo = (const float*)d_in[13];
  const float* ln1g = (const float*)d_in[14];
  const float* ln1b = (const float*)d_in[15];
  const float* W1 = (const float*)d_in[16];
  const float* b1 = (const float*)d_in[17];
  const float* W2 = (const float*)d_in[18];
  const float* b2 = (const float*)d_in[19];
  const float* ln2g = (const float*)d_in[20];
  const float* ln2b = (const float*)d_in[21];

  float* x = (float*)d_out;                       // fp32 activations live in d_out

  // workspace layout
  char* ws = (char*)d_ws;
  size_t off = 0;
  auto carve = [&](size_t bytes) { char* p = ws + off; off += (bytes + 255) & ~(size_t)255; return p; };
  const size_t actB = (size_t)MROWS * EMB;        // 16,711,680 elements
  __bf16* qb  = (__bf16*)carve(actB * 2);
  __bf16* kb  = (__bf16*)carve(actB * 2);
  __bf16* vb  = (__bf16*)carve(actB * 2);
  __bf16* vtb = (__bf16*)carve((size_t)BATCH * HEADS * DHEAD * SEQP * 2);
  __bf16* cb  = (__bf16*)carve(actB * 2);
  float*  tf  = (float*)carve(actB * 4);
  __bf16* xb  = (__bf16*)carve(actB * 2);
  __bf16* wts = (__bf16*)carve((size_t)NLAY * 786432 * 2);
  __bf16* hb  = (__bf16*)d_ws;                    // [M,1024] overlays qb..vtb (dead then)

  // 1. weight prep (fp32 -> bf16 transposed)
  for (int l = 0; l < NLAY; ++l) {
    __bf16* base = wts + (size_t)l * 786432;
    wprep_kernel<<<256, 256, 0, stream>>>(Wq + (size_t)l * 65536, base,           256, 256);
    wprep_kernel<<<256, 256, 0, stream>>>(Wk + (size_t)l * 65536, base + 65536,   256, 256);
    wprep_kernel<<<256, 256, 0, stream>>>(Wv + (size_t)l * 65536, base + 131072,  256, 256);
    wprep_kernel<<<256, 256, 0, stream>>>(Wo + (size_t)l * 65536, base + 196608,  256, 256);
    wprep_kernel<<<1024, 256, 0, stream>>>(W1 + (size_t)l * 262144, base + 262144, 256, 1024);
    wprep_kernel<<<1024, 256, 0, stream>>>(W2 + (size_t)l * 262144, base + 524288, 1024, 256);
  }

  // 2. embed: pad fill + visit scatter, then bf16 copy
  fill_pad_kernel<<<(unsigned)((actB + 255) / 256), 256, 0, stream>>>(x, pemb);
  visit_embed_kernel<<<BATCH * 128, 256, 0, stream>>>(codes, dest, times, cemb, x);
  cvt_bf16_kernel<<<(unsigned)((actB + 255) / 256), 256, 0, stream>>>(x, xb, actB);

  dim3 gProj(EMB / 128, MROWS / 128);     // (2, 510)
  dim3 gFF1(FFDIM / 128, MROWS / 128);    // (8, 510)
  size_t attnLds = (size_t)2 * 16 * SROW * sizeof(float);

  for (int l = 0; l < NLAY; ++l) {
    __bf16* Wqt = wts + (size_t)l * 786432;
    __bf16* Wkt = Wqt + 65536;
    __bf16* Wvt = Wqt + 131072;
    __bf16* Wot = Wqt + 196608;
    __bf16* W1t = Wqt + 262144;
    __bf16* W2t = Wqt + 524288;

    // QKV projections (bf16 out)
    gemm_bf16_kernel<<<gProj, 256, 0, stream>>>(xb, Wqt, bq + l * EMB, nullptr,
                                                nullptr, qb, MROWS, EMB, EMB, 0);
    gemm_bf16_kernel<<<gProj, 256, 0, stream>>>(xb, Wkt, bk + l * EMB, nullptr,
                                                nullptr, kb, MROWS, EMB, EMB, 0);
    gemm_bf16_kernel<<<gProj, 256, 0, stream>>>(xb, Wvt, bv + l * EMB, nullptr,
                                                nullptr, vb, MROWS, EMB, EMB, 0);
    // V transpose for K-contiguous B-fragments in ctx matmul
    vtrans_kernel<<<65536, 256, 0, stream>>>(vb, vtb);
    // attention
    attn_kernel<<<BATCH * HEADS * 16, 64, attnLds, stream>>>(qb, kb, vtb, cb);
    // O projection + residual, then LN1
    gemm_bf16_kernel<<<gProj, 256, 0, stream>>>(cb, Wot, bo + l * EMB, x,
                                                tf, nullptr, MROWS, EMB, EMB, 0);
    ln_kernel<<<MROWS / 8, 256, 0, stream>>>(tf, ln1g + l * EMB, ln1b + l * EMB, x, xb);
    // FFN
    gemm_bf16_kernel<<<gFF1, 256, 0, stream>>>(xb, W1t, b1 + l * FFDIM, nullptr,
                                               nullptr, hb, MROWS, FFDIM, EMB, 1);
    gemm_bf16_kernel<<<gProj, 256, 0, stream>>>(hb, W2t, b2 + l * EMB, x,
                                                tf, nullptr, MROWS, EMB, FFDIM, 0);
    ln_kernel<<<MROWS / 8, 256, 0, stream>>>(tf, ln2g + l * EMB, ln2b + l * EMB, x, xb);
  }
}
